// MeshCrossAttentionBlock_51316269252795
// MI455X (gfx1250) — matrix operand order repaired and checked by reading, not compiled, not run on hardware
//
#include <hip/hip_runtime.h>
#include <hip/hip_bf16.h>
#include <cstdint>

// ---------------------------------------------------------------------------
// MeshCrossAttentionBlock for MI455X (gfx1250, wave32, WMMA 16x16x32 f16)
// B=8, V=4096, L=1024, D=512, H=8, HD=64
// GEMMs use double-buffered async global->LDS copies (ASYNCcnt path).
// ---------------------------------------------------------------------------

typedef __attribute__((ext_vector_type(16))) _Float16 v16h;
typedef __attribute__((ext_vector_type(8)))  _Float16 v8h;
typedef __attribute__((ext_vector_type(8)))  float    v8f;

#define WMMA_F16(a, b, c) \
    __builtin_amdgcn_wmma_f32_16x16x32_f16(false, (a), false, (b), (short)0, (c), false, false)

// 16 contiguous halfs (16B-aligned start) -> v16h
static __device__ __forceinline__ v16h ld16h(const _Float16* p) {
    const v8h lo = *(const v8h*)p;
    const v8h hi = *(const v8h*)(p + 8);
    v16h r;
#pragma unroll
    for (int i = 0; i < 8; ++i) { r[i] = lo[i]; r[i + 8] = hi[i]; }
    return r;
}

// two separate 8-half chunks -> v16h (A-fragment gather)
static __device__ __forceinline__ v16h ld8x2h(const _Float16* p0, const _Float16* p1) {
    const v8h lo = *(const v8h*)p0;
    const v8h hi = *(const v8h*)(p1);
    v16h r;
#pragma unroll
    for (int i = 0; i < 8; ++i) { r[i] = lo[i]; r[i + 8] = hi[i]; }
    return r;
}

// Async 16B global->LDS copy (per-lane addresses), tracked by ASYNCcnt.
// LDS operand is the workgroup-relative byte offset = low 32 bits of the
// generic shared pointer (HW adds LDS_BASE).
static __device__ __forceinline__ void async_ld16B(const void* gptr, void* lptr) {
    unsigned loff = (unsigned)(uintptr_t)lptr;
    unsigned long long gaddr = (unsigned long long)(uintptr_t)gptr;
    asm volatile("global_load_async_to_lds_b128 %0, %1, off"
                 :: "v"(loff), "v"(gaddr)
                 : "memory");
}
template <int N>
static __device__ __forceinline__ void wait_asynccnt() {
    asm volatile("s_wait_asynccnt %0" :: "i"(N) : "memory");
}

// ---------------------------------------------------------------------------
// Weight convert + transpose: W [K][N] f32 -> WT [N][K] f16
// ---------------------------------------------------------------------------
__global__ __launch_bounds__(256) void convertT(const float* __restrict__ W,
                                                _Float16* __restrict__ WT, int K, int N) {
    int idx = blockIdx.x * 256 + threadIdx.x;
    if (idx >= K * N) return;
    int k = idx / N, n = idx % N;
    WT[(size_t)n * K + k] = (_Float16)W[idx];
}

// ---------------------------------------------------------------------------
// adaLN params: silu(cond) @ W_ada + b_ada  -> params [B][1024]
// ---------------------------------------------------------------------------
__global__ __launch_bounds__(256) void ada_kernel(const float* __restrict__ cond,
                                                  const float* __restrict__ W,
                                                  const float* __restrict__ bias,
                                                  float* __restrict__ params) {
    __shared__ float sl[512];
    const int b = blockIdx.x, t = threadIdx.x;
    for (int i = t; i < 512; i += 256) {
        float c = cond[b * 512 + i];
        sl[i] = c / (1.0f + __expf(-c));
    }
    __syncthreads();
#pragma unroll
    for (int j = 0; j < 4; ++j) {
        int n = t + j * 256;
        float acc = bias[n];
        for (int k = 0; k < 512; ++k) acc += sl[k] * W[k * 1024 + n];
        params[b * 1024 + n] = acc;
    }
}

// ---------------------------------------------------------------------------
// LayerNorm (D=512) + adaLN modulation: h = ln(x)*(1+scale[b]) + shift[b]  (f16)
// ---------------------------------------------------------------------------
__global__ __launch_bounds__(256) void ln_ada_kernel(const float* __restrict__ x,
                                                     const float* __restrict__ params,
                                                     _Float16* __restrict__ out,
                                                     int rowsPerBatch) {
    __shared__ float red[256];
    const int row = blockIdx.x, t = threadIdx.x;
    const int b = row / rowsPerBatch;
    const float* xr = x + (size_t)row * 512;
    float a0 = xr[t], a1 = xr[t + 256];

    red[t] = a0 + a1;
    __syncthreads();
    for (int off = 128; off > 0; off >>= 1) { if (t < off) red[t] += red[t + off]; __syncthreads(); }
    float mu = red[0] * (1.0f / 512.0f);
    __syncthreads();
    red[t] = a0 * a0 + a1 * a1;
    __syncthreads();
    for (int off = 128; off > 0; off >>= 1) { if (t < off) red[t] += red[t + off]; __syncthreads(); }
    float var = red[0] * (1.0f / 512.0f) - mu * mu;
    float inv = rsqrtf(var + 1e-5f);

    const float* sc = params + (size_t)b * 1024;
    const float* sh = sc + 512;
    out[(size_t)row * 512 + t]       = (_Float16)(((a0 - mu) * inv) * (1.0f + sc[t]) + sh[t]);
    out[(size_t)row * 512 + t + 256] = (_Float16)(((a1 - mu) * inv) * (1.0f + sc[t + 256]) + sh[t + 256]);
}

// ---------------------------------------------------------------------------
// LayerNorm + per-channel gamma/beta: ctx = ln(context)*g + b   (f16)
// ---------------------------------------------------------------------------
__global__ __launch_bounds__(256) void ln_gb_kernel(const float* __restrict__ x,
                                                    const float* __restrict__ g,
                                                    const float* __restrict__ bb,
                                                    _Float16* __restrict__ out) {
    __shared__ float red[256];
    const int row = blockIdx.x, t = threadIdx.x;
    const float* xr = x + (size_t)row * 512;
    float a0 = xr[t], a1 = xr[t + 256];

    red[t] = a0 + a1;
    __syncthreads();
    for (int off = 128; off > 0; off >>= 1) { if (t < off) red[t] += red[t + off]; __syncthreads(); }
    float mu = red[0] * (1.0f / 512.0f);
    __syncthreads();
    red[t] = a0 * a0 + a1 * a1;
    __syncthreads();
    for (int off = 128; off > 0; off >>= 1) { if (t < off) red[t] += red[t + off]; __syncthreads(); }
    float var = red[0] * (1.0f / 512.0f) - mu * mu;
    float inv = rsqrtf(var + 1e-5f);

    out[(size_t)row * 512 + t]       = (_Float16)(((a0 - mu) * inv) * g[t] + bb[t]);
    out[(size_t)row * 512 + t + 256] = (_Float16)(((a1 - mu) * inv) * g[t + 256] + bb[t + 256]);
}

// ---------------------------------------------------------------------------
// Generic WMMA GEMM: C[M,N] = A[M,K](f16, row-major) * BT[N,K](f16)^T
// block tile 64x64, 8 waves (4 m-tiles x 2 n-halves), K-step 32.
// Tiles staged with double-buffered async global->LDS copies.
// EPI 0: scatter to q16 [B][H][V][HD]
// EPI 1: scatter to k16 [B][H][L][HD] (n<512) and vT16 [B][H][HD][L] (n>=512)
// EPI 2: out = acc + bias[n] + resid[m][n] -> f32
// ---------------------------------------------------------------------------
template <int EPI>
__global__ __launch_bounds__(256) void gemm64(const _Float16* __restrict__ A,
                                              const _Float16* __restrict__ BT,
                                              int M, int N, int K,
                                              _Float16* __restrict__ o16a,
                                              _Float16* __restrict__ o16b,
                                              float* __restrict__ ofp,
                                              const float* __restrict__ bias,
                                              const float* __restrict__ resid,
                                              int RPB) {
    __shared__ __align__(16) _Float16 As[2][64 * 40];
    __shared__ __align__(16) _Float16 Bs[2][64 * 40];

    const int t = threadIdx.x;
    const int lane = t & 31, wv = t >> 5;
    const int wm = wv & 3, wn = wv >> 2;
    const int lm = lane & 15, lh = lane >> 4;
    const int m0 = blockIdx.y * 64, n0 = blockIdx.x * 64;

    const int lrow = t >> 2;        // 0..63
    const int lcol = (t & 3) * 8;   // 0,8,16,24
    const int lidx = lrow * 40 + lcol;

    const _Float16* gA = &A[(size_t)(m0 + lrow) * K + lcol];
    const _Float16* gB = &BT[(size_t)(n0 + lrow) * K + lcol];

    v8f c0, c1;
#pragma unroll
    for (int r = 0; r < 8; ++r) { c0[r] = 0.0f; c1[r] = 0.0f; }

    // prologue: async-copy first K-tile
    async_ld16B(gA, &As[0][lidx]);
    async_ld16B(gB, &Bs[0][lidx]);

    int cur = 0;
    for (int k0 = 0; k0 < K; k0 += 32) {
        if (k0 + 32 < K) {
            async_ld16B(gA + k0 + 32, &As[cur ^ 1][lidx]);
            async_ld16B(gB + k0 + 32, &Bs[cur ^ 1][lidx]);
            wait_asynccnt<2>();   // current tile's 2 copies complete (per wave)
        } else {
            wait_asynccnt<0>();
        }
        __syncthreads();

        const _Float16* ar = &As[cur][(wm * 16 + lm) * 40];
        v16h a = ld16h(ar + lh * 16);
        const _Float16* br0 = &Bs[cur][(wn * 32 + lm) * 40];
        const _Float16* br1 = &Bs[cur][(wn * 32 + 16 + lm) * 40];
        v16h b0 = ld16h(br0 + lh * 16);
        v16h b1 = ld16h(br1 + lh * 16);
        c0 = WMMA_F16(a, b0, c0);
        c1 = WMMA_F16(a, b1, c1);
        __syncthreads();
        cur ^= 1;
    }

    const int mbase = m0 + wm * 16 + lh * 8;
    const int col0 = n0 + wn * 32 + lm;
    const int col1 = col0 + 16;
#pragma unroll
    for (int r = 0; r < 8; ++r) {
        const int m = mbase + r;
        if constexpr (EPI == 0) {
            const int bb = m / RPB, vv = m % RPB;
            { int h = col0 >> 6, hd = col0 & 63;
              o16a[(((size_t)bb * 8 + h) * RPB + vv) * 64 + hd] = (_Float16)c0[r]; }
            { int h = col1 >> 6, hd = col1 & 63;
              o16a[(((size_t)bb * 8 + h) * RPB + vv) * 64 + hd] = (_Float16)c1[r]; }
        } else if constexpr (EPI == 1) {
            const int bb = m / RPB, ll = m % RPB;
            {   int n = col0;
                if (n < 512) { int h = n >> 6, hd = n & 63;
                    o16a[(((size_t)bb * 8 + h) * RPB + ll) * 64 + hd] = (_Float16)c0[r]; }
                else { int n2 = n - 512; int h = n2 >> 6, hd = n2 & 63;
                    o16b[(((size_t)bb * 8 + h) * 64 + hd) * (size_t)RPB + ll] = (_Float16)c0[r]; }
            }
            {   int n = col1;
                if (n < 512) { int h = n >> 6, hd = n & 63;
                    o16a[(((size_t)bb * 8 + h) * RPB + ll) * 64 + hd] = (_Float16)c1[r]; }
                else { int n2 = n - 512; int h = n2 >> 6, hd = n2 & 63;
                    o16b[(((size_t)bb * 8 + h) * 64 + hd) * (size_t)RPB + ll] = (_Float16)c1[r]; }
            }
        } else {
            ofp[(size_t)m * N + col0] = c0[r] + bias[col0] + resid[(size_t)m * N + col0];
            ofp[(size_t)m * N + col1] = c1[r] + bias[col1] + resid[(size_t)m * N + col1];
        }
    }
}

// ---------------------------------------------------------------------------
// Attention: per block -> one (b,h) and 16 query rows; full L=1024.
// Two-pass exact softmax: pass A (QK^T, online row max/sum), pass B
// (recompute scores -> f16 probs in LDS -> probs @ V via WMMA).
// 8 waves each own an L-chunk of 128.
// ---------------------------------------------------------------------------
__global__ __launch_bounds__(256) void attn_kernel(const _Float16* __restrict__ q16,
                                                   const _Float16* __restrict__ k16,
                                                   const _Float16* __restrict__ vT16,
                                                   const unsigned char* __restrict__ mask,
                                                   _Float16* __restrict__ ao16) {
    __shared__ __align__(16) _Float16 probs[8][16][136];
    __shared__ float outacc[16 * 64];
    __shared__ float statsM[8][16];
    __shared__ float statsS[8][16];
    __shared__ float rowM[16];
    __shared__ float rowS[16];

    const int t = threadIdx.x;
    const int lane = t & 31, w = t >> 5;
    const int lm = lane & 15, lh = lane >> 4;
    const int bh = blockIdx.y;          // b*8 + h
    const int b = bh >> 3, h = bh & 7;
    const int v0 = blockIdx.x * 16;

    for (int i = t; i < 16 * 64; i += 256) outacc[i] = 0.0f;

    // q A-fragments (16 rows x 64 K), K-steps 0 and 1
    const _Float16* qrow = q16 + (((size_t)bh) * 4096 + v0 + lm) * 64;
    const v16h aq0 = ld16h(qrow + lh * 16);
    const v16h aq1 = ld16h(qrow + 32 + lh * 16);

    const _Float16* kbase = k16 + ((size_t)bh * 1024) * 64;
    const unsigned char* mrow = mask + (size_t)b * 1024;
    const float scale = 0.125f;   // HD^-0.5
    const int chunk0 = w * 128;

    float rm[8], rs[8];
#pragma unroll
    for (int r = 0; r < 8; ++r) { rm[r] = -1e30f; rs[r] = 0.0f; }

    // ---- pass A: scores + online row stats ----
    for (int ti = 0; ti < 8; ++ti) {
        const int lc = chunk0 + ti * 16;
        const _Float16* krow = kbase + (size_t)(lc + lm) * 64;
        v16h b0 = ld16h(krow + lh * 16);
        v16h b1 = ld16h(krow + 32 + lh * 16);
        v8f c;
#pragma unroll
        for (int r = 0; r < 8; ++r) c[r] = 0.0f;
        c = WMMA_F16(aq0, b0, c);
        c = WMMA_F16(aq1, b1, c);
        const float mval = mrow[lc + lm] ? 0.0f : -1e30f;
#pragma unroll
        for (int r = 0; r < 8; ++r) {
            float s = c[r] * scale + mval;
            float mo = fmaxf(rm[r], s);
            rs[r] = rs[r] * __expf(rm[r] - mo) + __expf(s - mo);
            rm[r] = mo;
        }
    }
    // merge across the 16 lanes of each half-wave (same rows, different cols)
#pragma unroll
    for (int r = 0; r < 8; ++r) {
        float m = rm[r], s = rs[r];
#pragma unroll
        for (int off = 1; off < 16; off <<= 1) {
            float m2 = __shfl_xor(m, off, 32);
            float s2 = __shfl_xor(s, off, 32);
            float mo = fmaxf(m, m2);
            s = s * __expf(m - mo) + s2 * __expf(m2 - mo);
            m = mo;
        }
        rm[r] = m; rs[r] = s;
    }
    if (lm == 0) {
#pragma unroll
        for (int r = 0; r < 8; ++r) { statsM[w][lh * 8 + r] = rm[r]; statsS[w][lh * 8 + r] = rs[r]; }
    }
    __syncthreads();
    if (t < 16) {
        float m = statsM[0][t], s = statsS[0][t];
        for (int ww = 1; ww < 8; ++ww) {
            float m2 = statsM[ww][t], s2 = statsS[ww][t];
            float mo = fmaxf(m, m2);
            s = s * __expf(m - mo) + s2 * __expf(m2 - mo);
            m = mo;
        }
        rowM[t] = m;
        rowS[t] = 1.0f / fmaxf(s, 1e-30f);
    }
    __syncthreads();

    float rmx[8], rsi[8];
#pragma unroll
    for (int r = 0; r < 8; ++r) { rmx[r] = rowM[lh * 8 + r]; rsi[r] = rowS[lh * 8 + r]; }

    // ---- pass B: recompute scores -> f16 probabilities in LDS ----
    for (int ti = 0; ti < 8; ++ti) {
        const int lc = chunk0 + ti * 16;
        const _Float16* krow = kbase + (size_t)(lc + lm) * 64;
        v16h b0 = ld16h(krow + lh * 16);
        v16h b1 = ld16h(krow + 32 + lh * 16);
        v8f c;
#pragma unroll
        for (int r = 0; r < 8; ++r) c[r] = 0.0f;
        c = WMMA_F16(aq0, b0, c);
        c = WMMA_F16(aq1, b1, c);
        const float mval = mrow[lc + lm] ? 0.0f : -1e30f;
#pragma unroll
        for (int r = 0; r < 8; ++r) {
            float p = __expf(c[r] * scale + mval - rmx[r]) * rsi[r];
            probs[w][lh * 8 + r][ti * 16 + lm] = (_Float16)p;
        }
    }
    __syncthreads();

    // ---- probs(16x128) @ V(128x64) via WMMA over this wave's chunk ----
    const _Float16* vbase = vT16 + ((size_t)bh * 64) * 1024;
    v8f o[4];
#pragma unroll
    for (int nt = 0; nt < 4; ++nt)
#pragma unroll
        for (int r = 0; r < 8; ++r) o[nt][r] = 0.0f;

#pragma unroll
    for (int kk = 0; kk < 4; ++kk) {
        const _Float16* pr = &probs[w][lm][0];
        v16h ap = ld16h(pr + kk * 32 + lh * 16);
#pragma unroll
        for (int nt = 0; nt < 4; ++nt) {
            const _Float16* vrow = vbase + (size_t)(nt * 16 + lm) * 1024 + chunk0;
            v16h bv = ld16h(vrow + kk * 32 + lh * 16);
            o[nt] = WMMA_F16(ap, bv, o[nt]);
        }
    }
    // reduce partial outputs across the 8 waves
#pragma unroll
    for (int nt = 0; nt < 4; ++nt) {
        const int col = nt * 16 + lm;
#pragma unroll
        for (int r = 0; r < 8; ++r) atomicAdd(&outacc[(lh * 8 + r) * 64 + col], o[nt][r]);
    }
    __syncthreads();

    for (int i = t; i < 16 * 64; i += 256) {
        const int row = i >> 6, hd = i & 63;
        ao16[((size_t)b * 4096 + v0 + row) * 512 + h * 64 + hd] = (_Float16)outacc[i];
    }
}

// ---------------------------------------------------------------------------
// host launcher
// ---------------------------------------------------------------------------
extern "C" void kernel_launch(void* const* d_in, const int* in_sizes, int n_in,
                              void* d_out, int out_size, void* d_ws, size_t ws_size,
                              hipStream_t stream) {
    (void)in_sizes; (void)n_in; (void)out_size; (void)ws_size;

    const float* x     = (const float*)d_in[0];
    const float* ctx   = (const float*)d_in[1];
    const float* cond  = (const float*)d_in[2];
    const unsigned char* cmask = (const unsigned char*)d_in[3];
    const float* W_ada = (const float*)d_in[4];
    const float* b_ada = (const float*)d_in[5];
    const float* g_ctx = (const float*)d_in[6];
    const float* b_ctx = (const float*)d_in[7];
    const float* Wq    = (const float*)d_in[8];
    const float* Wkv   = (const float*)d_in[9];
    const float* Wp    = (const float*)d_in[10];
    const float* bp    = (const float*)d_in[11];
    float* out = (float*)d_out;

    char* ws = (char*)d_ws;
    size_t off = 0;
    auto alloc = [&](size_t bytes) -> char* {
        char* p = ws + off;
        off += (bytes + 255) & ~(size_t)255;
        return p;
    };
    float*    params = (float*)alloc((size_t)8 * 1024 * 4);
    _Float16* WqT    = (_Float16*)alloc((size_t)512 * 512 * 2);
    _Float16* WkvT   = (_Float16*)alloc((size_t)1024 * 512 * 2);
    _Float16* WpT    = (_Float16*)alloc((size_t)512 * 512 * 2);
    _Float16* h16    = (_Float16*)alloc((size_t)32768 * 512 * 2);
    _Float16* c16    = (_Float16*)alloc((size_t)8192 * 512 * 2);
    _Float16* q16    = (_Float16*)alloc((size_t)64 * 4096 * 64 * 2);
    _Float16* k16    = (_Float16*)alloc((size_t)64 * 1024 * 64 * 2);
    _Float16* vT16   = (_Float16*)alloc((size_t)64 * 64 * 1024 * 2);
    _Float16* ao16   = (_Float16*)alloc((size_t)32768 * 512 * 2);

    convertT<<<(512 * 512 + 255) / 256, 256, 0, stream>>>(Wq, WqT, 512, 512);
    convertT<<<(512 * 1024 + 255) / 256, 256, 0, stream>>>(Wkv, WkvT, 512, 1024);
    convertT<<<(512 * 512 + 255) / 256, 256, 0, stream>>>(Wp, WpT, 512, 512);

    ada_kernel<<<8, 256, 0, stream>>>(cond, W_ada, b_ada, params);
    ln_ada_kernel<<<32768, 256, 0, stream>>>(x, params, h16, 4096);
    ln_gb_kernel<<<8192, 256, 0, stream>>>(ctx, g_ctx, b_ctx, c16);

    // q = h @ Wq  (M=32768, N=512, K=512) -> q16 [B][H][V][HD]
    gemm64<0><<<dim3(512 / 64, 32768 / 64), 256, 0, stream>>>(
        h16, WqT, 32768, 512, 512, q16, nullptr, nullptr, nullptr, nullptr, 4096);
    // kv = ctx @ Wkv (M=8192, N=1024, K=512) -> k16 / vT16
    gemm64<1><<<dim3(1024 / 64, 8192 / 64), 256, 0, stream>>>(
        c16, WkvT, 8192, 1024, 512, k16, vT16, nullptr, nullptr, nullptr, 1024);

    attn_kernel<<<dim3(4096 / 16, 64), 256, 0, stream>>>(q16, k16, vT16, cmask, ao16);

    // out = x + ao @ Wp + bp  (M=32768, N=512, K=512) -> f32
    gemm64<2><<<dim3(512 / 64, 32768 / 64), 256, 0, stream>>>(
        ao16, WpT, 32768, 512, 512, nullptr, nullptr, out, bp, x, 0);
}
